// ManualAttention1D_63290638074523
// MI455X (gfx1250) — matrix operand order repaired and checked by reading, not compiled, not run on hardware
//
#include <hip/hip_runtime.h>

typedef __attribute__((ext_vector_type(16))) _Float16 v16h;
typedef __attribute__((ext_vector_type(8)))  float    v8f;

#define CCH 512
#define BLK 256
#define LOG2E 1.44269504088896340736f

__global__ __launch_bounds__(BLK) void attn1d_rank1_wmma(
    const float* __restrict__ qg, const float* __restrict__ kg, const float* __restrict__ vg,
    const float* __restrict__ wqg, const float* __restrict__ wkg, const float* __restrict__ wvg,
    float* __restrict__ outg)
{
    __shared__ __align__(16) float sq[CCH];
    __shared__ __align__(16) float sk[CCH];
    __shared__ __align__(16) float sv[CCH];
    __shared__ __align__(16) float qh2[CCH];     // conv(q) * log2(e)
    __shared__ __align__(16) float khs[CCH];     // conv(k)
    __shared__ __align__(32) _Float16 vhh[CCH];  // conv(v) as f16 (B-matrix col 0)
    __shared__ __align__(16) float numb[CCH];
    __shared__ __align__(16) float denb[CCH];
    __shared__ float redmax[BLK];
    __shared__ float redmin[BLK];
    __shared__ float s_khmax, s_khmin;

    const int tid = threadIdx.x;
    const int bat = blockIdx.x;
    const float* qb = qg + bat * CCH;
    const float* kb = kg + bat * CCH;
    const float* vb = vg + bat * CCH;

    // ---- stage raw inputs to LDS ----
    sq[tid]       = qb[tid];
    sq[tid + 256] = qb[tid + 256];
    sk[tid]       = kb[tid];
    sk[tid + 256] = kb[tid + 256];
    sv[tid]       = vb[tid];
    sv[tid + 256] = vb[tid + 256];
    float cwq[5], cwk[5], cwv[5];
#pragma unroll
    for (int j = 0; j < 5; ++j) { cwq[j] = wqg[j]; cwk[j] = wkg[j]; cwv[j] = wvg[j]; }
    __syncthreads();

    // ---- 5-tap same-padded convs (correlation, matching lax.conv) ----
    float lmax = -1e30f, lmin = 1e30f;
#pragma unroll
    for (int r = 0; r < 2; ++r) {
        const int c = tid + r * 256;
        float aq = 0.f, ak = 0.f, av = 0.f;
#pragma unroll
        for (int j = 0; j < 5; ++j) {
            const int idx = c - 2 + j;
            if (idx >= 0 && idx < CCH) {
                aq = fmaf(sq[idx], cwq[j], aq);
                ak = fmaf(sk[idx], cwk[j], ak);
                av = fmaf(sv[idx], cwv[j], av);
            }
        }
        qh2[c] = aq * LOG2E;
        khs[c] = ak;
        vhh[c] = (_Float16)av;
        lmax = fmaxf(lmax, ak);
        lmin = fminf(lmin, ak);
    }
    redmax[tid] = lmax;
    redmin[tid] = lmin;
    __syncthreads();
    for (int s = BLK / 2; s > 0; s >>= 1) {
        if (tid < s) {
            redmax[tid] = fmaxf(redmax[tid], redmax[tid + s]);
            redmin[tid] = fminf(redmin[tid], redmin[tid + s]);
        }
        __syncthreads();
    }
    if (tid == 0) { s_khmax = redmax[0]; s_khmin = redmin[0]; }
    __syncthreads();
    const float khmax = s_khmax;
    const float khmin = s_khmin;

    // ---- main exp + WMMA reduction ----
    const int lane   = tid & 31;
    const int wid    = tid >> 5;          // 8 waves
    const int ncol   = lane & 15;         // A row / B,C,D column index
    const int hi     = (lane >> 4) & 1;
    const int kbA    = hi ? 16 : 0;       // A k order used here (contiguous, not ISA table order): lanes 0-15 K0-15, lanes 16-31 K16-31
    const int kbB    = hi ? 16 : 0;       // B k order used here (contiguous, same as A): lanes 0-15 K0-15, lanes 16-31 K16-31
    const int rbase  = hi ? 8  : 0;       // C/D: lanes 0-15 rows 0-7, lanes 16-31 rows 8-15

    float qc[4], m2[4];
    v8f acc[4];
#pragma unroll
    for (int i = 0; i < 4; ++i) {
        const int c = (wid * 4 + i) * 16 + ncol;
        const float qv = qh2[c];
        qc[i] = qv;
        m2[i] = (qv >= 0.f) ? qv * khmax * LOG2E / LOG2E : qv * khmin; // m2 in log2 domain below
        m2[i] = (qv >= 0.f) ? qv * khmax : qv * khmin;                 // qv already has log2e folded in
        v8f z = {};
        acc[i] = z;
    }

    v16h onesv, zerosv;
#pragma unroll
    for (int j = 0; j < 16; ++j) { onesv[j] = (_Float16)1.0f; zerosv[j] = (_Float16)0.0f; }

    for (int chunk = 0; chunk < 16; ++chunk) {
        const int D0 = chunk * 32;

        // B tile: col0 = vh, col1 = ones, rest zero (shared across the 4 row-tiles)
        const v16h bload = *reinterpret_cast<const v16h*>(&vhh[D0 + kbB]);
        const v16h Bm = (ncol == 0) ? bload : ((ncol == 1) ? onesv : zerosv);

        // kh values for this lane's 16 K slots (broadcast LDS b128 loads)
        float khv[16];
        {
            const float4 a0 = *reinterpret_cast<const float4*>(&khs[D0 + kbA +  0]);
            const float4 a1 = *reinterpret_cast<const float4*>(&khs[D0 + kbA +  4]);
            const float4 a2 = *reinterpret_cast<const float4*>(&khs[D0 + kbA +  8]);
            const float4 a3 = *reinterpret_cast<const float4*>(&khs[D0 + kbA + 12]);
            khv[0]=a0.x; khv[1]=a0.y; khv[2]=a0.z; khv[3]=a0.w;
            khv[4]=a1.x; khv[5]=a1.y; khv[6]=a1.z; khv[7]=a1.w;
            khv[8]=a2.x; khv[9]=a2.y; khv[10]=a2.z; khv[11]=a2.w;
            khv[12]=a3.x; khv[13]=a3.y; khv[14]=a3.z; khv[15]=a3.w;
        }

#pragma unroll
        for (int i = 0; i < 4; ++i) {
            v16h A;
#pragma unroll
            for (int j = 0; j < 16; ++j) {
                // P = 2^(qh*log2e * kh - m2) in (0,1] : one v_fma_f32 + one v_exp_f32
                const float e = __builtin_amdgcn_exp2f(fmaf(qc[i], khv[j], -m2[i]));
                A[j] = (_Float16)e;
            }
            acc[i] = __builtin_amdgcn_wmma_f32_16x16x32_f16(
                false, A, false, Bm, (short)0, acc[i], false, false);
        }
    }

    // ---- scatter num (col 0) / den (col 1) and divide ----
#pragma unroll
    for (int i = 0; i < 4; ++i) {
        const int cb = (wid * 4 + i) * 16 + rbase;
        if (ncol == 0) {
#pragma unroll
            for (int r = 0; r < 8; ++r) numb[cb + r] = acc[i][r];
        } else if (ncol == 1) {
#pragma unroll
            for (int r = 0; r < 8; ++r) denb[cb + r] = acc[i][r];
        }
    }
    __syncthreads();
#pragma unroll
    for (int r = 0; r < 2; ++r) {
        const int c = tid + r * 256;
        outg[bat * CCH + c] = numb[c] / denb[c];
    }
}

extern "C" void kernel_launch(void* const* d_in, const int* in_sizes, int n_in,
                              void* d_out, int out_size, void* d_ws, size_t ws_size,
                              hipStream_t stream) {
    const float* q  = (const float*)d_in[0];
    const float* k  = (const float*)d_in[1];
    const float* v  = (const float*)d_in[2];
    const float* wq = (const float*)d_in[3];
    const float* wk = (const float*)d_in[4];
    const float* wv = (const float*)d_in[5];
    float* out = (float*)d_out;

    const int B = in_sizes[0] / CCH;   // 256
    attn1d_rank1_wmma<<<B, BLK, 0, stream>>>(q, k, v, wq, wk, wv, out);
    (void)n_in; (void)out_size; (void)d_ws; (void)ws_size;
}
